// PPDFM_18863496364187
// MI455X (gfx1250) — matrix operand order repaired and checked by reading, not compiled, not run on hardware
//
#include <hip/hip_runtime.h>
#include <hip/hip_bf16.h>
#include <cstddef>

// ============================================================================
// PPDFM on MI455X (gfx1250): implicit-GEMM convs via V_WMMA_F32_16X16X32_F16.
// Compute-bound (~0.6 TFLOP/call; all tensors L2-resident, 23.3 TB/s HBM is
// never the limiter) -> push convs through the WMMA pipe, f16 in / f32 acc.
// Block tile 64px x 64cout, 2 WMMAs per wave per K-step sharing one B frag,
// im2col index decode hoisted into a per-block LDS table.
// ============================================================================

typedef __attribute__((ext_vector_type(16))) _Float16 v16h;
typedef __attribute__((ext_vector_type(8)))  _Float16 v8h;
typedef __attribute__((ext_vector_type(8)))  float    v8f;

#define ACT_NONE  0
#define ACT_LRELU 1
#define KMAX      4384   // largest Kpad (omc: 486*9=4374 -> 4384)

// ---------------------------------------------------------------------------
// Weight packing: OIHW f32 -> [Npad rows][Kpad] f16, zero padded.
// Row o, entry k = c*kh*kw + ky*kw + kx (contiguous in source), so each
// lane's B-fragment (N = row, K contiguous) is two 16B reads.
// ---------------------------------------------------------------------------
__global__ void prep_w_kernel(const float* __restrict__ w, _Float16* __restrict__ wt,
                              int Cout, int K, int Kpad, int total) {
  int i = blockIdx.x * 256 + threadIdx.x;
  if (i >= total) return;
  int o = i / Kpad;
  int k = i - o * Kpad;
  float v = (o < Cout && k < K) ? w[(size_t)o * K + k] : 0.f;
  wt[i] = (_Float16)v;
}

// ---------------------------------------------------------------------------
// Implicit-GEMM conv.
//   Block = 256 threads = 8 waves; tile = 64 px (M) x 64 cout (N).
//   Wave (mg = wave&1, ns = wave>>1): pixels [mg*32, mg*32+32) as two 16-px
//   A fragments, couts [ns*16, ns*16+16) as one B fragment -> 2 wmma/K-step.
//   A frag layout: lane<16 -> M=lane, K{0..7,16..23}; lane>=16 -> K{8..15,24..31}.
//   B frag layout: lane n = lane&15, same K split as A (K{0..7,16..23} / K{8..15,24..31}).
//   Per-block LDS k-decode table kills the div/mod in the hot loop.
//   Epilogue fuses bias + up to two residual adds + leaky-relu.
// ---------------------------------------------------------------------------
__global__ __launch_bounds__(256)
void conv_wmma_kernel(const float* __restrict__ in,
                      const _Float16* __restrict__ wt,
                      const float* __restrict__ bias,
                      const float* __restrict__ add0,
                      const float* __restrict__ add1,
                      float* __restrict__ out,
                      int Cin, int Cout, int Npad, int Kpad,
                      int inH, int inW, int outH, int outW,
                      int stride, int ks, int act) {
  __shared__ __align__(16) _Float16 At[64 * 32];   // 4 KB A tile
  __shared__ int   pyx[64 * 2];                    // per-pixel base coords
  __shared__ int   koff[KMAX];                     // c*inHW + dy*inW + dx
  __shared__ short kdydx[KMAX];                    // (dy+1) | (dx+1)<<8 ; 0x7F7F = dead

  const int tid  = threadIdx.x;
  const int wave = tid >> 5;
  const int lane = tid & 31;
  const int mg   = wave & 1;        // which 32-px half
  const int ns   = wave >> 1;       // which 16-cout column
  const int HWo  = outH * outW;
  const int inHW = inH * inW;
  const int pixBase = blockIdx.x * 64;
  const int nbase   = blockIdx.y * 64 + ns * 16;
  const bool nact   = (nbase < Npad);
  const int K       = Cin * ks * ks;
  const int kBlocks = Kpad >> 5;

  // ---- one-time: pixel coords + k-decode table (covered by first sync) ----
  if (tid < 64) {
    int p = pixBase + tid;
    if (p < HWo) {
      int py = p / outW;
      int px = p - py * outW;
      pyx[2 * tid]     = py * stride;
      pyx[2 * tid + 1] = px * stride;
    } else {
      pyx[2 * tid]     = -100000;
      pyx[2 * tid + 1] = -100000;
    }
  }
  for (int k = tid; k < Kpad; k += 256) {
    if (k < K) {
      int c, dy, dx;
      if (ks == 3) {
        c = k / 9;
        int rr = k - c * 9;
        dy = rr / 3;
        dx = rr - dy * 3;       // dy,dx in 0..2 (offset +1 form)
      } else {
        c = k; dy = 1; dx = 1;
      }
      koff[k]  = c * inHW + (dy - 1) * inW + (dx - 1);
      kdydx[k] = (short)(dy | (dx << 8));
    } else {
      koff[k]  = 0;
      kdydx[k] = (short)0x7F7F;   // shifts y,x by +126; B is 0 for k >= K
    }
  }

  const _Float16* wrow =
      wt + (size_t)(nbase + (lane & 15)) * Kpad + ((lane >> 4) << 3);

  v8f acc0 = {};
  v8f acc1 = {};

  for (int kb = 0; kb < kBlocks; ++kb) {
    __syncthreads();
    // ---- im2col fill: each thread stages 8 consecutive K as one b128 ----
    {
      const int pl = tid >> 2;          // pixel 0..63
      const int k0 = (tid & 3) << 3;    // K {0,8,16,24}
      const int iy = pyx[2 * pl];
      const int ix = pyx[2 * pl + 1];
      const int base = iy * inW + ix;
      _Float16 tmp[8];
#pragma unroll
      for (int j = 0; j < 8; ++j) {
        const int k = (kb << 5) + k0 + j;
        const int pk = kdydx[k];
        const int y = iy + (pk & 0xff) - 1;
        const int x = ix + (pk >> 8) - 1;
        float v = 0.f;
        if ((unsigned)y < (unsigned)inH && (unsigned)x < (unsigned)inW)
          v = in[(size_t)(base + koff[k])];
        tmp[j] = (_Float16)v;
      }
      *(v8h*)&At[(pl << 5) + k0] = *(v8h*)tmp;
    }
    __syncthreads();
    // ---- 2 WMMAs per wave, shared B fragment ----
    {
      const int r0 = (mg << 5) + (lane & 15);       // frag0 row
      const int hi = (lane >> 4) << 3;              // 0 or 8
      const v8h a0lo = *(const v8h*)&At[(r0 << 5) + hi];
      const v8h a0hi = *(const v8h*)&At[(r0 << 5) + 16 + hi];
      const v8h a1lo = *(const v8h*)&At[((r0 + 16) << 5) + hi];
      const v8h a1hi = *(const v8h*)&At[((r0 + 16) << 5) + 16 + hi];
      if (nact) {
        const v8h b0 = *(const v8h*)(wrow + (kb << 5));
        const v8h b1 = *(const v8h*)(wrow + (kb << 5) + 16);
        v16h A0, A1, B;
#pragma unroll
        for (int e = 0; e < 8; ++e) {
          A0[e] = a0lo[e]; A0[e + 8] = a0hi[e];
          A1[e] = a1lo[e]; A1[e + 8] = a1hi[e];
          B[e]  = b0[e];   B[e + 8]  = b1[e];
        }
        acc0 = __builtin_amdgcn_wmma_f32_16x16x32_f16(
            false, A0, false, B, (short)0, acc0, false, false);
        acc1 = __builtin_amdgcn_wmma_f32_16x16x32_f16(
            false, A1, false, B, (short)0, acc1, false, false);
        if (kb + 1 < kBlocks)
          __builtin_prefetch(wrow + ((kb + 1) << 5), 0, 1);
      }
    }
  }

  // ---- epilogue: C layout -> lane n = lane&15, m = (lane>>4)*8 + r ----
  if (nact) {
    const int n = nbase + (lane & 15);
    if (n < Cout) {
      const float bco = bias ? bias[n] : 0.f;
      const size_t cbase = (size_t)n * HWo;
      const int pm0 = pixBase + mg * 32 + (lane >> 4) * 8;
#pragma unroll
      for (int f = 0; f < 2; ++f) {
        const v8f* accp = f ? &acc1 : &acc0;
        const int pm = pm0 + f * 16;
#pragma unroll
        for (int r = 0; r < 8; ++r) {
          const int p = pm + r;
          if (p < HWo) {
            float v = (*accp)[r] + bco;
            if (add0) v += add0[cbase + p];
            if (add1) v += add1[cbase + p];
            if (act == ACT_LRELU) v = (v >= 0.f) ? v : 0.2f * v;
            out[cbase + p] = v;
          }
        }
      }
    }
  }
}

// ---------------------------------------------------------------------------
// Deformable conv (G=9 groups, cpg=1, K=9, Cin=9, Cout=64) with fused
// sigmoid mask + bias + lrelu. One thread per pixel, 64 f32 accumulators.
// om layout: ch ((g*9+k)*2+d) = offsets, ch 162+g*9+k = mask logits.
// ---------------------------------------------------------------------------
__global__ __launch_bounds__(256)
void deform_kernel(const float* __restrict__ x, const float* __restrict__ om,
                   const float* __restrict__ w, const float* __restrict__ b,
                   float* __restrict__ out, int H, int Wd) {
  const int p  = blockIdx.x * 256 + threadIdx.x;
  const int HW = H * Wd;
  if (p >= HW) return;
  const int py0 = p / Wd;
  const int px0 = p - py0 * Wd;

  float acc[64];
#pragma unroll
  for (int o = 0; o < 64; ++o) acc[o] = 0.f;

  for (int g = 0; g < 9; ++g) {
    const float* xc = x + (size_t)g * HW;
    for (int k = 0; k < 9; ++k) {
      const int ky = k / 3 - 1;
      const int kx = k - (k / 3) * 3 - 1;
      const int ch = g * 9 + k;
      const float oy = om[(size_t)(ch * 2 + 0) * HW + p];
      const float ox = om[(size_t)(ch * 2 + 1) * HW + p];
      float m = om[(size_t)(162 + ch) * HW + p];
      m = 1.f / (1.f + __expf(-m));
      const float fy = (float)(py0 + ky) + oy;
      const float fx = (float)(px0 + kx) + ox;
      const float y0f = floorf(fy), x0f = floorf(fx);
      const float wy = fy - y0f, wx = fx - x0f;
      const int y0 = (int)y0f, x0 = (int)x0f;

      auto samp = [&](int yi, int xi) -> float {
        const bool valid = (yi >= 0) && (yi < H) && (xi >= 0) && (xi < Wd);
        const int yq = min(max(yi, 0), H - 1);
        const int xq = min(max(xi, 0), Wd - 1);
        const float t = xc[(size_t)yq * Wd + xq];
        return valid ? t : 0.f;
      };
      const float v00 = samp(y0,     x0);
      const float v01 = samp(y0,     x0 + 1);
      const float v10 = samp(y0 + 1, x0);
      const float v11 = samp(y0 + 1, x0 + 1);
      float v = v00 * (1.f - wy) * (1.f - wx) + v01 * (1.f - wy) * wx +
                v10 * wy * (1.f - wx) + v11 * wy * wx;
      v *= m;
#pragma unroll
      for (int o = 0; o < 64; ++o)
        acc[o] = fmaf(v, w[(size_t)o * 81 + ch], acc[o]);
    }
  }
#pragma unroll
  for (int o = 0; o < 64; ++o) {
    float r = acc[o] + b[o];
    out[(size_t)o * HW + p] = (r >= 0.f) ? r : 0.2f * r;
  }
}

// ---------------------------------------------------------------------------
// Bilinear 2x upsample (half-pixel centers, edge clamp) with scale multiplier.
// ---------------------------------------------------------------------------
__global__ void up2_kernel(const float* __restrict__ in, float* __restrict__ out,
                           int C, int H, int Wd, float mult, int total) {
  int idx = blockIdx.x * 256 + threadIdx.x;
  if (idx >= total) return;
  const int Ho = 2 * H, Wo = 2 * Wd;
  const int c  = idx / (Ho * Wo);
  const int r  = idx - c * Ho * Wo;
  const int oy = r / Wo;
  const int ox = r - oy * Wo;
  const float fy = oy * 0.5f - 0.25f;
  const float fx = ox * 0.5f - 0.25f;
  const float y0f = floorf(fy), x0f = floorf(fx);
  const float wy = fy - y0f, wx = fx - x0f;
  int y0 = (int)y0f, x0 = (int)x0f;
  int y1 = y0 + 1, x1 = x0 + 1;
  y0 = min(max(y0, 0), H - 1);  y1 = min(max(y1, 0), H - 1);
  x0 = min(max(x0, 0), Wd - 1); x1 = min(max(x1, 0), Wd - 1);
  const float* ic = in + (size_t)c * H * Wd;
  const float v =
      (1.f - wy) * ((1.f - wx) * ic[(size_t)y0 * Wd + x0] + wx * ic[(size_t)y0 * Wd + x1]) +
      wy        * ((1.f - wx) * ic[(size_t)y1 * Wd + x0] + wx * ic[(size_t)y1 * Wd + x1]);
  out[idx] = v * mult;
}

// ===========================================================================
// Host-side orchestration
// ===========================================================================

namespace {

struct Arena {
  char*  base;
  size_t off, cap;
  void* alloc_bytes(size_t n) {
    size_t a = (off + 255) & ~(size_t)255;
    off = a + n;
    return (void*)(base + a);
  }
  float* allocf(size_t n) { return (float*)alloc_bytes(n * sizeof(float)); }
};

struct ConvW {
  const _Float16* wt;
  const float* bias;
  int Cin, Cout, ks, Kpad, Npad;
};

struct PdfbW {
  ConvW l1, l21, l22, l31, l32, catoff, rb[6], om, omc, ff;
  const float *dc_w, *dc_b;
};

struct PpdfbW {
  ConvW conv0, merge, dense0, dense1, dense2, c1x1;
  PdfbW pdfb;
};

struct Ws {
  float *cat1, *cat2, *cat3, *sa, *sb;
  float *ofA, *ofB, *rbT;
  float *om, *omc_in;
  float *ff, *ff_cat;
  float *fused, *spa[3], *merge_in, *dbuf;
  float *xa[3], *xb[3];
};

static ConvW prep_conv(Arena& ar, hipStream_t s, const float* w, const float* b,
                       int Cout, int Cin, int ks) {
  ConvW cw;
  cw.Cin = Cin; cw.Cout = Cout; cw.ks = ks; cw.bias = b;
  const int K = Cin * ks * ks;
  cw.Kpad = (K + 31) & ~31;
  cw.Npad = (Cout + 15) & ~15;
  _Float16* wt = (_Float16*)ar.alloc_bytes((size_t)cw.Kpad * cw.Npad * sizeof(_Float16));
  const int total = cw.Kpad * cw.Npad;
  prep_w_kernel<<<(total + 255) / 256, 256, 0, s>>>(w, wt, Cout, K, cw.Kpad, total);
  cw.wt = wt;
  return cw;
}

static void conv(hipStream_t s, const float* in, const ConvW& w, float* out,
                 int inH, int inW, int stride, int act,
                 const float* add0 = nullptr, const float* add1 = nullptr) {
  const int outH = (w.ks == 3) ? inH / stride : inH;
  const int outW = (w.ks == 3) ? inW / stride : inW;
  dim3 grid((outH * outW + 63) / 64, (w.Npad + 63) / 64);
  conv_wmma_kernel<<<grid, 256, 0, s>>>(in, w.wt, w.bias, add0, add1, out,
                                        w.Cin, w.Cout, w.Npad, w.Kpad,
                                        inH, inW, outH, outW, stride, w.ks, act);
}

static void up2(hipStream_t s, const float* in, float* out, int C, int H, int Wd,
                float mult) {
  const int total = C * 4 * H * Wd;
  up2_kernel<<<(total + 255) / 256, 256, 0, s>>>(in, out, C, H, Wd, mult, total);
}

static PdfbW prep_pdfb(Arena& ar, hipStream_t s, const float* const* L) {
  // Leaf order = dict insertion order of init_pdfb():
  // l1, l2_1, l2_2, l3_1, l3_2, cat_off, rb[0..2].{w1,w2},
  // om_w, om_b, dc_w, dc_b, omc_w, omc_b, ff_w, ff_b
  PdfbW P;
  P.l1     = prep_conv(ar, s, L[0],  nullptr, 3,  64, 3);
  P.l21    = prep_conv(ar, s, L[1],  nullptr, 3,  3,  3);
  P.l22    = prep_conv(ar, s, L[2],  nullptr, 3,  3,  3);
  P.l31    = prep_conv(ar, s, L[3],  nullptr, 3,  3,  3);
  P.l32    = prep_conv(ar, s, L[4],  nullptr, 3,  3,  3);
  P.catoff = prep_conv(ar, s, L[5],  nullptr, 64, 9,  3);
  for (int r = 0; r < 6; ++r)
    P.rb[r] = prep_conv(ar, s, L[6 + r], nullptr, 64, 64, 3);
  P.om   = prep_conv(ar, s, L[12], L[13], 243, 64,  3);
  P.dc_w = L[14];
  P.dc_b = L[15];
  P.omc  = prep_conv(ar, s, L[16], L[17], 243, 486, 3);
  P.ff   = prep_conv(ar, s, L[18], L[19], 64,  128, 3);
  return P;
}

static void run_pdfb(hipStream_t s, Ws& W, const PdfbW& P,
                     const float* x0, const float* x1, const float* x2,
                     float* outbuf) {
  const int H1 = 192, W1 = 192, H2 = 96, W2 = 96, H3 = 48, W3 = 48;
  const int HW1 = H1 * W1, HW2 = H2 * W2, HW3 = H3 * W3;
  const float* xs[3] = {x0, x1, x2};

  // 3-scale feature pyramid, written directly into concat slices
  for (int j = 0; j < 3; ++j) {
    float* f1 = W.cat1 + (size_t)j * 3 * HW1;
    conv(s, xs[j], P.l1, f1, H1, W1, 1, ACT_LRELU);
    conv(s, f1, P.l21, W.sa, H1, W1, 2, ACT_LRELU);
    float* f2 = W.cat2 + (size_t)j * 3 * HW2;
    conv(s, W.sa, P.l22, f2, H2, W2, 1, ACT_LRELU);
    conv(s, f2, P.l31, W.sb, H2, W2, 2, ACT_LRELU);
    float* f3 = W.cat3 + (size_t)j * 3 * HW3;
    conv(s, W.sb, P.l32, f3, H3, W3, 1, ACT_LRELU);
  }

  for (int i = 3; i >= 1; --i) {
    const int Hh = (i == 1) ? H1 : (i == 2) ? H2 : H3;
    const int Ww = Hh;
    const int HW = Hh * Ww;
    const float* cf = (i == 1) ? W.cat1 : (i == 2) ? W.cat2 : W.cat3;

    // of = conv(cf, cat_off); then 3 residual blocks (ping-pong)
    conv(s, cf, P.catoff, W.ofA, Hh, Ww, 1, ACT_NONE);
    float* ofc = W.ofA;
    float* ofn = W.ofB;
    for (int r = 0; r < 3; ++r) {
      conv(s, ofc, P.rb[2 * r], W.rbT, Hh, Ww, 1, ACT_LRELU);
      conv(s, W.rbT, P.rb[2 * r + 1], ofn, Hh, Ww, 1, ACT_NONE, ofc);
      float* t = ofc; ofc = ofn; ofn = t;
    }

    // om (243ch); at i<3 concat with up2(prev om)*2 and refine with omc
    const float* omsrc;
    if (i == 3) {
      conv(s, ofc, P.om, W.om, Hh, Ww, 1, ACT_NONE);
      omsrc = W.om;
    } else {
      conv(s, ofc, P.om, W.omc_in, Hh, Ww, 1, ACT_NONE);   // ch 0..242
      // ch 243..485 hold up2(prev om)*2, written at the previous scale
      conv(s, W.omc_in, P.omc, W.om, Hh, Ww, 1, ACT_NONE);
      omsrc = W.om;
    }

    // ff = lrelu(deform(cf, off, sigmoid(msk))); at i<3 concat + ff conv
    float* ffdst = (i == 3) ? W.ff : W.ff_cat;   // ch 0..63 of concat buffer
    deform_kernel<<<(HW + 255) / 256, 256, 0, s>>>(cf, omsrc, P.dc_w, P.dc_b,
                                                   ffdst, Hh, Ww);
    const float* ffcur;
    if (i == 3) {
      ffcur = W.ff;
    } else {
      float* tgt = (i == 1) ? outbuf : W.ff;
      conv(s, W.ff_cat, P.ff, tgt, Hh, Ww, 1, ACT_NONE);
      ffcur = tgt;
    }

    if (i > 1) {
      const int HWn = HW * 4;  // next (finer) scale pixel count
      up2(s, omsrc, W.omc_in + (size_t)243 * HWn, 243, Hh, Ww, 2.0f);
      up2(s, ffcur, W.ff_cat + (size_t)64 * HWn, 64, Hh, Ww, 1.0f);
    }
  }
}

static void run_ppdfb(hipStream_t s, Ws& W, const PpdfbW& B,
                      const float* const xin[3], float* const xout[3]) {
  const int H = 192, Wd = 192;
  const size_t HW = (size_t)H * Wd;

  for (int j = 0; j < 3; ++j)
    conv(s, xin[j], B.conv0, W.spa[j], H, Wd, 1, ACT_LRELU);

  run_pdfb(s, W, B.pdfb, W.spa[0], W.spa[1], W.spa[2], W.fused);

  hipMemcpyAsync(W.merge_in, W.fused, 64 * HW * sizeof(float),
                 hipMemcpyDeviceToDevice, s);
  for (int j = 0; j < 3; ++j) {
    hipMemcpyAsync(W.merge_in + 64 * HW, W.spa[j], 64 * HW * sizeof(float),
                   hipMemcpyDeviceToDevice, s);
    conv(s, W.merge_in, B.merge, W.dbuf, H, Wd, 1, ACT_LRELU);  // r -> D[0:64]
    conv(s, W.dbuf, B.dense0, W.dbuf + 64 * HW,  H, Wd, 1, ACT_LRELU);
    conv(s, W.dbuf, B.dense1, W.dbuf + 128 * HW, H, Wd, 1, ACT_LRELU);
    conv(s, W.dbuf, B.dense2, W.dbuf + 192 * HW, H, Wd, 1, ACT_LRELU);
    // out = conv1x1(D[0:256]) + r + xs[j]   (both adds fused in epilogue)
    conv(s, W.dbuf, B.c1x1, xout[j], H, Wd, 1, ACT_NONE, W.dbuf, xin[j]);
  }
}

}  // namespace

extern "C" void kernel_launch(void* const* d_in, const int* in_sizes, int n_in,
                              void* d_out, int out_size, void* d_ws, size_t ws_size,
                              hipStream_t stream) {
  (void)in_sizes; (void)out_size;
  if (n_in < 49) return;

  // Input index map: setup_inputs() dict insertion order, recursively:
  //   0..2   : x0, x1, x2
  //   3      : ppdfb.conv0
  //   4..23  : ppdfb.pdfb (init_pdfb insertion order, 20 leaves)
  //   24     : ppdfb.merge
  //   25..27 : ppdfb.rdb.dense[0..2]
  //   28     : ppdfb.rdb.c1x1
  //   29..48 : pdfb (init_pdfb insertion order, 20 leaves)
  const float* IN[64];
  for (int i = 0; i < n_in && i < 64; ++i) IN[i] = (const float*)d_in[i];

  Arena ar;
  ar.base = (char*)d_ws;
  ar.off  = 0;
  ar.cap  = ws_size;

  // ---- pack all conv weights to f16 (once per launch; deterministic) ----
  PpdfbW B;
  B.conv0  = prep_conv(ar, stream, IN[3],  nullptr, 64, 64, 3);
  B.pdfb   = prep_pdfb(ar, stream, &IN[4]);
  B.merge  = prep_conv(ar, stream, IN[24], nullptr, 64, 128, 1);
  B.dense0 = prep_conv(ar, stream, IN[25], nullptr, 64, 64, 3);
  B.dense1 = prep_conv(ar, stream, IN[26], nullptr, 64, 128, 3);
  B.dense2 = prep_conv(ar, stream, IN[27], nullptr, 64, 192, 3);
  B.c1x1   = prep_conv(ar, stream, IN[28], nullptr, 64, 256, 1);
  PdfbW Pfinal = prep_pdfb(ar, stream, &IN[29]);

  // ---- activation arena (~330 MB) ----
  const size_t HW1 = 192 * 192, HW2 = 96 * 96, HW3 = 48 * 48;
  Ws W;
  W.cat1     = ar.allocf(9 * HW1);
  W.cat2     = ar.allocf(9 * HW2);
  W.cat3     = ar.allocf(9 * HW3);
  W.sa       = ar.allocf(3 * HW1);
  W.sb       = ar.allocf(3 * HW1);
  W.ofA      = ar.allocf(64 * HW1);
  W.ofB      = ar.allocf(64 * HW1);
  W.rbT      = ar.allocf(64 * HW1);
  W.om       = ar.allocf(243 * HW1);
  W.omc_in   = ar.allocf(486 * HW1);
  W.ff       = ar.allocf(64 * HW1);
  W.ff_cat   = ar.allocf(128 * HW1);
  W.fused    = ar.allocf(64 * HW1);
  for (int j = 0; j < 3; ++j) W.spa[j] = ar.allocf(64 * HW1);
  W.merge_in = ar.allocf(128 * HW1);
  W.dbuf     = ar.allocf(256 * HW1);
  for (int j = 0; j < 3; ++j) W.xa[j] = ar.allocf(64 * HW1);
  for (int j = 0; j < 3; ++j) W.xb[j] = ar.allocf(64 * HW1);

  // ---- 3x PPDFB (same params each iteration, per the reference) ----
  const float* it1_in[3] = {IN[0], IN[1], IN[2]};
  run_ppdfb(stream, W, B, it1_in, W.xa);
  const float* it2_in[3] = {W.xa[0], W.xa[1], W.xa[2]};
  run_ppdfb(stream, W, B, it2_in, W.xb);
  const float* it3_in[3] = {W.xb[0], W.xb[1], W.xb[2]};
  run_ppdfb(stream, W, B, it3_in, W.xa);

  // ---- final PDFB -> d_out (64 x 192 x 192 fp32) ----
  run_pdfb(stream, W, Pfinal, W.xa[0], W.xa[1], W.xa[2], (float*)d_out);
}
